// EGCL_22497038697192
// MI455X (gfx1250) — compile-verified, matrix operand order checked by reading, not run on hardware
//
#include <hip/hip_runtime.h>
#include <hip/hip_bf16.h>

typedef __attribute__((ext_vector_type(16))) _Float16 v16h;
typedef __attribute__((ext_vector_type(8)))  _Float16 v8h;
typedef __attribute__((ext_vector_type(8)))  float    v8f;

#define HD   64     // node feature dim
#define INK  128    // layer-1 GEMM K (2*HD); radial handled as rank-1 update
#define HID  256    // hidden dim
#define TE   32     // edges per block-iteration (two 16-row WMMA tiles)

// Fast SiLU: x * rcp(1 + exp(-x)).  v_rcp_f32 instead of IEEE-div sequence.
__device__ __forceinline__ float silu_f(float x) {
    return x * __builtin_amdgcn_rcpf(1.0f + __expf(-x));
}

// Load one v16h WMMA fragment (A or B) given the per-lane row base pointer.
// 16-bit 16x32 layout: halves [kb..kb+7] -> elems 0..7, [kb+16..kb+23] -> 8..15.
__device__ __forceinline__ v16h load_frag16(const _Float16* p) {
    v8h lo = *(const v8h*)(p);
    v8h hi = *(const v8h*)(p + 16);
    v16h r;
#pragma unroll
    for (int i = 0; i < 8; ++i) { r[i] = lo[i]; r[i + 8] = hi[i]; }
    return r;
}

// ---------------------------------------------------------------------------
// Prep: transpose + f32->f16 convert weights so a B fragment is two
// contiguous 16B loads per lane (W^T row-major).
// ---------------------------------------------------------------------------
__global__ void egcl_prep(const float* __restrict__ W1,
                          const float* __restrict__ W2,
                          const float* __restrict__ W3,
                          _Float16* __restrict__ wt1,   // [256][128]
                          _Float16* __restrict__ wt2,   // [256][256]
                          _Float16* __restrict__ wt3,   // [256][256]
                          float* __restrict__ w1last) { // [256] = W1 row 128
    int i = blockIdx.x * blockDim.x + threadIdx.x;
    if (i < HID * INK) {
        int n = i >> 7, k = i & (INK - 1);
        wt1[i] = (_Float16)W1[k * HID + n];
    }
    if (i < HID * HID) {
        int n = i >> 8, k = i & (HID - 1);
        wt2[i] = (_Float16)W2[k * HID + n];
        wt3[i] = (_Float16)W3[k * HID + n];
    }
    if (i < HID) w1last[i] = W1[INK * HID + i];
}

__global__ void egcl_zero(float* __restrict__ p, int n) {
    int i = blockIdx.x * blockDim.x + threadIdx.x;
    if (i < n) p[i] = 0.0f;
}

// ---------------------------------------------------------------------------
// Fused edge MLP: 512 threads (16 wave32s). Wave w owns hidden columns
// [16w, 16w+16); weight fragments are VGPR-resident across the edge loop.
// Each iteration processes 32 edges = two 16-row tiles with independent
// accumulator chains (hides WMMA D->C latency, halves barrier overhead).
// ---------------------------------------------------------------------------
__launch_bounds__(512, 1)
__global__ void egcl_main(const float* __restrict__ h,
                          const float* __restrict__ cdiff,
                          const long long* __restrict__ ei,   // [2][E] int64
                          const float* __restrict__ b1,
                          const float* __restrict__ b2,
                          const float* __restrict__ b3,
                          const float* __restrict__ W4,       // [256]
                          const _Float16* __restrict__ wt1,
                          const _Float16* __restrict__ wt2,
                          const _Float16* __restrict__ wt3,
                          const float* __restrict__ w1last,
                          float* __restrict__ seg,            // [N][3]
                          float* __restrict__ cnt,            // [N]
                          int E) {
    __shared__ _Float16 actA[TE][HID];   // 16 KB (layer1 in / layer2 out)
    __shared__ _Float16 actB[TE][HID];   // 16 KB (layer1 out / layer2 in)
    __shared__ float    rad[TE];
    __shared__ float    sbuf[TE];

    const int tid  = threadIdx.x;
    const int lane = tid & 31;
    const int wave = tid >> 5;
    const int n    = (wave << 4) + (lane & 15);   // this lane's hidden column
    const int hi8  = (lane >> 4) << 3;            // 0 or 8 (k-half / row-half)
    const int m0   = lane & 15;                   // A-frag row, tile 0
    const int m1   = 16 + m0;                     // A-frag row, tile 1

    // --- resident weight fragments: 20 x v16h = 160 VGPRs/lane ---
    v16h bw1[4], bw2[8], bw3[8];
#pragma unroll
    for (int c = 0; c < 4; ++c) bw1[c] = load_frag16(wt1 + n * INK + 32 * c + hi8);
#pragma unroll
    for (int c = 0; c < 8; ++c) bw2[c] = load_frag16(wt2 + n * HID + 32 * c + hi8);
#pragma unroll
    for (int c = 0; c < 8; ++c) bw3[c] = load_frag16(wt3 + n * HID + 32 * c + hi8);

    const float bias1 = b1[n], bias2 = b2[n], bias3 = b3[n];
    const float wl = w1last[n], w4v = W4[n];

    const v8f accb1 = {bias1, bias1, bias1, bias1, bias1, bias1, bias1, bias1};
    const v8f accb2 = {bias2, bias2, bias2, bias2, bias2, bias2, bias2, bias2};
    const v8f accb3 = {bias3, bias3, bias3, bias3, bias3, bias3, bias3, bias3};

    const int nTiles = (E + TE - 1) / TE;
    for (int t = blockIdx.x; t < nTiles; t += gridDim.x) {
        const int e0 = t * TE;

        // ---- gather 32 edges: cat(h[row], h[col]) -> f16 LDS [32][128] ----
        {
            int e = tid >> 4, q = tid & 15;        // 8 floats per thread
            int ge = e0 + e;
            float4 va = make_float4(0.f, 0.f, 0.f, 0.f);
            float4 vb = va;
            if (ge < E) {
                long long r  = ei[ge];
                long long c2 = ei[E + ge];
                const float* src = (q < 8) ? (h + r * HD + 8 * q)
                                           : (h + c2 * HD + (8 * q - HD));
                va = *(const float4*)src;
                vb = *(const float4*)(src + 4);
            }
            v8h hv;
            hv[0] = (_Float16)va.x; hv[1] = (_Float16)va.y;
            hv[2] = (_Float16)va.z; hv[3] = (_Float16)va.w;
            hv[4] = (_Float16)vb.x; hv[5] = (_Float16)vb.y;
            hv[6] = (_Float16)vb.z; hv[7] = (_Float16)vb.w;
            *(v8h*)&actA[e][8 * q] = hv;
        }
        if (tid < TE) {
            int ge = e0 + tid;
            float rr = 0.f;
            if (ge < E) {
                float a = cdiff[ge * 3 + 0];
                float b = cdiff[ge * 3 + 1];
                float c = cdiff[ge * 3 + 2];
                rr = a * a + b * b + c * c;
            }
            rad[tid]  = rr;
            sbuf[tid] = 0.f;
        }
        __syncthreads();

        // ---- layer 1: K=128 GEMM + bias(init) + rank-1 radial, SiLU ----
        {
            v8f acc0 = accb1, acc1 = accb1;
#pragma unroll
            for (int c = 0; c < 4; ++c) {
                v16h a0 = load_frag16(&actA[m0][32 * c + hi8]);
                v16h a1 = load_frag16(&actA[m1][32 * c + hi8]);
                acc0 = __builtin_amdgcn_wmma_f32_16x16x32_f16(
                    false, a0, false, bw1[c], (short)0, acc0, false, false);
                acc1 = __builtin_amdgcn_wmma_f32_16x16x32_f16(
                    false, a1, false, bw1[c], (short)0, acc1, false, false);
            }
#pragma unroll
            for (int v = 0; v < 8; ++v) {     // C rows: v+hi8 and 16+v+hi8
                float x0 = acc0[v] + rad[hi8 + v] * wl;
                float x1 = acc1[v] + rad[16 + hi8 + v] * wl;
                actB[hi8 + v][n]      = (_Float16)silu_f(x0);
                actB[16 + hi8 + v][n] = (_Float16)silu_f(x1);
            }
        }
        __syncthreads();

        // ---- layer 2: K=256 GEMM + bias(init), SiLU ----
        {
            v8f acc0 = accb2, acc1 = accb2;
#pragma unroll
            for (int c = 0; c < 8; ++c) {
                v16h a0 = load_frag16(&actB[m0][32 * c + hi8]);
                v16h a1 = load_frag16(&actB[m1][32 * c + hi8]);
                acc0 = __builtin_amdgcn_wmma_f32_16x16x32_f16(
                    false, a0, false, bw2[c], (short)0, acc0, false, false);
                acc1 = __builtin_amdgcn_wmma_f32_16x16x32_f16(
                    false, a1, false, bw2[c], (short)0, acc1, false, false);
            }
#pragma unroll
            for (int v = 0; v < 8; ++v) {
                actA[hi8 + v][n]      = (_Float16)silu_f(acc0[v]);
                actA[16 + hi8 + v][n] = (_Float16)silu_f(acc1[v]);
            }
        }
        __syncthreads();

        // ---- layer 3: K=256 GEMM + bias(init), SiLU, fused @W4 + reduce ----
        {
            v8f acc0 = accb3, acc1 = accb3;
#pragma unroll
            for (int c = 0; c < 8; ++c) {
                v16h a0 = load_frag16(&actA[m0][32 * c + hi8]);
                v16h a1 = load_frag16(&actA[m1][32 * c + hi8]);
                acc0 = __builtin_amdgcn_wmma_f32_16x16x32_f16(
                    false, a0, false, bw3[c], (short)0, acc0, false, false);
                acc1 = __builtin_amdgcn_wmma_f32_16x16x32_f16(
                    false, a1, false, bw3[c], (short)0, acc1, false, false);
            }
            float p0[8], p1[8];
#pragma unroll
            for (int v = 0; v < 8; ++v) {
                p0[v] = silu_f(acc0[v]) * w4v;
                p1[v] = silu_f(acc1[v]) * w4v;
            }
            // butterfly sum over the 16 columns this wave owns
#pragma unroll
            for (int m = 1; m <= 8; m <<= 1) {
#pragma unroll
                for (int v = 0; v < 8; ++v) {
                    p0[v] += __shfl_xor(p0[v], m, 32);
                    p1[v] += __shfl_xor(p1[v], m, 32);
                }
            }
            if ((lane & 15) == 0) {   // lane 0 -> rows 0-7, lane 16 -> rows 8-15
#pragma unroll
                for (int v = 0; v < 8; ++v) {
                    atomicAdd(&sbuf[hi8 + v], p0[v]);
                    atomicAdd(&sbuf[16 + hi8 + v], p1[v]);
                }
            }
        }
        __syncthreads();

        // ---- scatter: trans = clip(coord_diff * s) -> segment sums ----
        if (tid < TE) {
            int ge = e0 + tid;
            if (ge < E) {
                float s = sbuf[tid];
                long long r = ei[ge];
#pragma unroll
                for (int d = 0; d < 3; ++d) {
                    float tr = cdiff[ge * 3 + d] * s;
                    tr = fminf(fmaxf(tr, -100.0f), 100.0f);
                    atomicAdd(&seg[r * 3 + d], tr);
                }
                atomicAdd(&cnt[r], 1.0f);
            }
        }
        __syncthreads();   // protects sbuf/actA reuse next iteration
    }
}

__global__ void egcl_finalize(const float* __restrict__ seg,
                              const float* __restrict__ cnt,
                              float* __restrict__ out, int N) {
    int i = blockIdx.x * blockDim.x + threadIdx.x;
    if (i < N) {
        float inv = 1.0f / fmaxf(cnt[i], 1.0f);
        out[3 * i + 0] = seg[3 * i + 0] * inv;
        out[3 * i + 1] = seg[3 * i + 1] * inv;
        out[3 * i + 2] = seg[3 * i + 2] * inv;
    }
}

// ---------------------------------------------------------------------------
extern "C" void kernel_launch(void* const* d_in, const int* in_sizes, int n_in,
                              void* d_out, int out_size, void* d_ws, size_t ws_size,
                              hipStream_t stream) {
    const float*     h  = (const float*)d_in[0];
    const float*     cd = (const float*)d_in[1];
    const long long* ei = (const long long*)d_in[2];
    const float*     W1 = (const float*)d_in[3];
    const float*     b1 = (const float*)d_in[4];
    const float*     W2 = (const float*)d_in[5];
    const float*     b2 = (const float*)d_in[6];
    const float*     W3 = (const float*)d_in[7];
    const float*     b3 = (const float*)d_in[8];
    const float*     W4 = (const float*)d_in[9];

    const int N = in_sizes[0] / HD;   // 50000
    const int E = in_sizes[1] / 3;    // 800000

    char* ws = (char*)d_ws;
    const size_t OFF_WT1 = 0;                              // 256*128*2 = 64 KB
    const size_t OFF_WT2 = OFF_WT1 + (size_t)HID * INK * 2;
    const size_t OFF_WT3 = OFF_WT2 + (size_t)HID * HID * 2;
    const size_t OFF_W1L = OFF_WT3 + (size_t)HID * HID * 2;
    const size_t OFF_SEG = OFF_W1L + (size_t)HID * 4;      // seg[N*3] then cnt[N]
    const size_t OFF_CNT = OFF_SEG + (size_t)N * 3 * 4;

    _Float16* wt1 = (_Float16*)(ws + OFF_WT1);
    _Float16* wt2 = (_Float16*)(ws + OFF_WT2);
    _Float16* wt3 = (_Float16*)(ws + OFF_WT3);
    float* w1last = (float*)(ws + OFF_W1L);
    float* seg    = (float*)(ws + OFF_SEG);
    float* cnt    = (float*)(ws + OFF_CNT);

    // weight transpose/convert
    egcl_prep<<<(HID * HID + 255) / 256, 256, 0, stream>>>(
        W1, W2, W3, wt1, wt2, wt3, w1last);
    // zero seg+cnt (contiguous 4N floats)
    egcl_zero<<<(4 * N + 255) / 256, 256, 0, stream>>>(seg, 4 * N);
    // fused edge MLP + scatter
    egcl_main<<<1536, 512, 0, stream>>>(h, cd, ei, b1, b2, b3, W4,
                                        wt1, wt2, wt3, w1last, seg, cnt, E);
    // segment mean
    egcl_finalize<<<(N + 255) / 256, 256, 0, stream>>>(seg, cnt, (float*)d_out, N);
}
